// GCNLayer_18760417148942
// MI455X (gfx1250) — matrix operand order repaired and checked by reading, not compiled, not run on hardware
//
#include <hip/hip_runtime.h>

// GCN layer on MI455X (gfx1250).
// Roofline: adjacency = 1 GiB -> ~46us floor @ 23.3 TB/s. A.T@h = 8.6 GFLOP
// needs ~187 TFLOP/s to hide under the stream -> WMMA pipe required.
// A in {0,1} is exact in f16; h->f16 with f32 accumulate (v_wmma_f32_16x16x32_f16).
// Little's law: 23.3 TB/s * ~500ns => ~12 MB in flight. 4096 blocks (8 K-slices
// x 512 col-tiles) x 2 waves x 16 outstanding NT loads x 128B ~= 16 MB, via
// TLP (depth-1 per-wave prefetch) -- the compiler renames this shape cleanly.

#define NNODES 16384
#define IN_F   128
#define H1     15
#define PSI_F  16
#define FH     25
#define OUT_F  128
#define NCHUNK (NNODES / 32)       // K-chunks of 32 for 16x16x32 WMMA
#define SLICES 8                   // K split across blocks (deterministic partials)
#define SPAN   (NCHUNK / SLICES)   // 64 chunks per block (even)

typedef __attribute__((ext_vector_type(16))) _Float16 v16h;
typedef __attribute__((ext_vector_type(8)))  float    v8f;

union FragU { unsigned int w[8]; uint4 q[2]; v16h v; };

// Workgroup barrier that orders LDS traffic only: drains DScnt, NOT LOADcnt,
// so prefetched global loads stay in flight across the tile handoff.
// (__syncthreads lowers to s_wait_loadcnt_dscnt 0x0 which kills the pipeline.)
__device__ __forceinline__ void lds_barrier() {
    asm volatile(
        "s_wait_dscnt 0x0\n\t"
        "s_barrier_signal -1\n\t"
        "s_barrier_wait -1"
        ::: "memory");
}

// ---------------------------------------------------------------------------
// Kernel 1: h = relu(relu(nodes@psi_w1+b1)@psi_w2+b2), written as f16 in the
// WMMA B-operand (32x16, K=j, N=f) layout:
//   dword (chunk*32 + L)*8 + v  holds { h[32c + 2v + 16*(L/16)][L%16],
//                                       h[32c + 2v+1 + 16*(L/16)][L%16] }
// so aggregation lanes fetch their B fragment as 2 contiguous b128 loads.
// ---------------------------------------------------------------------------
__global__ void __launch_bounds__(64) psi_mlp_kernel(
    const float* __restrict__ nodes,
    const float* __restrict__ w1, const float* __restrict__ b1,
    const float* __restrict__ w2, const float* __restrict__ b2,
    unsigned short* __restrict__ hwB)
{
    __shared__ float sW1[IN_F * H1];
    __shared__ float sB1[H1];
    __shared__ float sW2[H1 * PSI_F];
    __shared__ float sB2[PSI_F];
    __shared__ float sX[64 * 130];     // 64 node rows, pad 130 -> conflict-free

    const int t  = threadIdx.x;
    const int n0 = blockIdx.x * 64;

    for (int i = t; i < IN_F * H1; i += 64) sW1[i] = w1[i];
    for (int i = t; i < H1 * PSI_F; i += 64) sW2[i] = w2[i];
    if (t < H1)    sB1[t] = b1[t];
    if (t < PSI_F) sB2[t] = b2[t];
    for (int i = t; i < 64 * IN_F; i += 64) {          // coalesced stage
        int r = i >> 7, c = i & 127;
        sX[r * 130 + c] = nodes[(n0 + r) * IN_F + c];
    }
    __syncthreads();

    float a1[H1];
    #pragma unroll
    for (int j = 0; j < H1; ++j) a1[j] = sB1[j];
    for (int k = 0; k < IN_F; ++k) {
        float xk = sX[t * 130 + k];
        #pragma unroll
        for (int j = 0; j < H1; ++j) a1[j] = fmaf(xk, sW1[k * H1 + j], a1[j]);
    }
    #pragma unroll
    for (int j = 0; j < H1; ++j) a1[j] = fmaxf(a1[j], 0.0f);

    const int n      = n0 + t;
    const int chunk  = n >> 5;
    const int r      = n & 31;
    const int Lbase  = (r >> 4) << 4;   // lanes 16-31 hold K=16..31
    const int v      = (r & 15) >> 1;   // dword within fragment
    const int halfIx = r & 1;           // low/high f16 in dword
    #pragma unroll
    for (int f = 0; f < PSI_F; ++f) {
        float hv = sB2[f];
        #pragma unroll
        for (int j = 0; j < H1; ++j) hv = fmaf(a1[j], sW2[j * PSI_F + f], hv);
        hv = fmaxf(hv, 0.0f);
        _Float16 hh = (_Float16)hv;
        int L = f + Lbase;
        hwB[((((chunk * 32 + L) * 8) + v) << 1) + halfIx] =
            __builtin_bit_cast(unsigned short, hh);
    }
}

// ---------------------------------------------------------------------------
// Kernel 2: per-slice partials of psi = A.T @ h (WMMA) and counts = colsum(A).
// Grid (512, SLICES) x 64 threads (2 waves). Block owns 32 columns x 1/SLICES
// of K. Double-buffered LDS pkA, stride 20 dwords (80B) -> 16B-aligned,
// bank-conflict-free b128:
//   pkA[col][k] = { f16 A[2k][col], f16 A[2k+1][col] }  (K row-pairs per dword)
// A-frag, contiguous K: lane<16 -> K 0-15 (k 0-7),
//                        lane>=16 -> K 16-31 (k 8-15).
// f16 pack via integer math: (a0 | a1<<16) * 0x3C00 == {f16(a0), f16(a1)}.
// Loop shape = av + nv prefetch + tail copy: compiler unrolls x2 and renames
// the copies away (verified in round-2 codegen; manual rotation caused
// v_mov churn + loadcnt-0 drains).
// ---------------------------------------------------------------------------
__global__ void __launch_bounds__(64) aggregate_kernel(
    const int* __restrict__ A,
    const unsigned int* __restrict__ hwB32,
    float* __restrict__ psi_part,          // [SLICES][NNODES*PSI_F]
    unsigned int* __restrict__ cnt_part)   // [SLICES][NNODES]
{
    __shared__ __align__(16) unsigned int pkA[2][32 * 20];
    __shared__ unsigned int cbuf[64];

    const int t     = threadIdx.x;
    const int i0    = blockIdx.x * 32;
    const int slice = blockIdx.y;
    const int c0    = slice * SPAN;
    const int col   = t & 31;          // column this thread loads
    const int half  = t >> 5;          // K-half this thread loads == wave id
    const int lane  = t & 31;
    const int mycol = half * 16 + (lane & 15);            // wave's WMMA M col
    const int abase = mycol * 20 + ((lane >= 16) ? 8 : 0);
    const int wslot = col * 20 + half * 8;

    unsigned int cnt = 0;
    v8f acc = {};

    // depth-1 register prefetch: preload chunk 0
    int av[16];
    {
        const int* Ap = A + (c0 * 32 + half * 16) * NNODES + i0 + col;
        #pragma unroll
        for (int e = 0; e < 16; ++e)
            av[e] = __builtin_nontemporal_load(Ap + e * NNODES);
    }

    for (int chunk = 0; chunk < SPAN; ++chunk) {
        // 1. B fragment for this chunk (L2-resident stream) -- issued first
        FragU b;
        {
            const uint4* bp =
                (const uint4*)(hwB32 + (c0 + chunk) * 256 + lane * 8);
            b.q[0] = bp[0];
            b.q[1] = bp[1];
        }

        // 2. issue next chunk's HBM loads; they stay in flight across the
        //    pack, the LDS-only barrier, and the WMMA (precise loadcnt waits)
        int nc = (chunk + 1 < SPAN) ? (chunk + 1) : chunk;
        int nv[16];
        {
            const int* Ap = A + ((c0 + nc) * 32 + half * 16) * NNODES + i0 + col;
            #pragma unroll
            for (int e = 0; e < 16; ++e)
                nv[e] = __builtin_nontemporal_load(Ap + e * NNODES);
        }

        // 3. integer-pack current chunk to f16 pairs, accumulate counts
        FragU pa;
        #pragma unroll
        for (int k = 0; k < 8; ++k) {
            unsigned int a0 = (unsigned int)av[2 * k];
            unsigned int a1 = (unsigned int)av[2 * k + 1];
            cnt += a0 + a1;
            pa.w[k] = (a0 | (a1 << 16)) * 0x3C00u;
        }

        unsigned int* pb = pkA[chunk & 1];
        *(uint4*)&pb[wslot]     = pa.q[0];
        *(uint4*)&pb[wslot + 4] = pa.q[1];
        lds_barrier();     // DScnt-only barrier; double buffering makes the
                           // read->next-write direction safe (reads are
                           // dscnt-drained before each wave's wmma, which
                           // precedes its next signal)

        FragU a;
        a.q[0] = *(const uint4*)&pb[abase];
        a.q[1] = *(const uint4*)&pb[abase + 4];

        acc = __builtin_amdgcn_wmma_f32_16x16x32_f16(
            false, a.v, false, b.v, (short)0, acc, false, false);

        #pragma unroll
        for (int e = 0; e < 16; ++e) av[e] = nv[e];
    }

    // D layout: lane holds N = lane%16, M = r + 8*(lane/16) for VGPR r
    const int fcol = lane & 15;
    const int mrow = i0 + half * 16 + ((lane >> 4) << 3);
    float* pout = psi_part + (size_t)slice * NNODES * PSI_F;
    #pragma unroll
    for (int rI = 0; rI < 8; ++rI)
        pout[(mrow + rI) * PSI_F + fcol] = acc[rI];

    cbuf[t] = cnt;
    __syncthreads();
    if (t < 32) cnt_part[slice * NNODES + i0 + t] = cbuf[t] + cbuf[t + 32];
}

// ---------------------------------------------------------------------------
// Kernel 3: out = relu(relu([nodes, (sum_s psi_part)/(sum_s cnt)] @ fi_w1 + b1)
//                      @ fi_w2 + b2)   -- fixed-order partial reduction.
// ---------------------------------------------------------------------------
__global__ void __launch_bounds__(64) fi_mlp_kernel(
    const float* __restrict__ nodes,
    const float* __restrict__ psi_part,
    const unsigned int* __restrict__ cnt_part,
    const float* __restrict__ w1, const float* __restrict__ b1,
    const float* __restrict__ w2, const float* __restrict__ b2,
    float* __restrict__ out)
{
    __shared__ float sW1[(IN_F + PSI_F) * FH];   // 144*25
    __shared__ float sB1[FH];
    __shared__ float sW2[FH * OUT_F];            // 25*128
    __shared__ float sB2[OUT_F];
    __shared__ float sG[64 * (FH + 1)];          // pad 26

    const int t  = threadIdx.x;
    const int n0 = blockIdx.x * 64;

    for (int i = t; i < (IN_F + PSI_F) * FH; i += 64) sW1[i] = w1[i];
    for (int i = t; i < FH * OUT_F; i += 64) sW2[i] = w2[i];
    if (t < FH) sB1[t] = b1[t];
    for (int i = t; i < OUT_F; i += 64) sB2[i] = b2[i];
    __syncthreads();

    const int n = n0 + t;
    float g[FH];
    #pragma unroll
    for (int j = 0; j < FH; ++j) g[j] = sB1[j];

    const float* xrow = nodes + n * IN_F;
    for (int k = 0; k < IN_F; ++k) {
        float zk = xrow[k];
        #pragma unroll
        for (int j = 0; j < FH; ++j) g[j] = fmaf(zk, sW1[k * FH + j], g[j]);
    }

    // reduce K-slice partials (deterministic order)
    unsigned int cn = 0;
    float ps[PSI_F];
    #pragma unroll
    for (int k = 0; k < PSI_F; ++k) ps[k] = 0.0f;
    #pragma unroll
    for (int s = 0; s < SLICES; ++s) {
        cn += cnt_part[s * NNODES + n];
        #pragma unroll
        for (int k = 0; k < PSI_F; ++k)
            ps[k] += psi_part[(size_t)s * NNODES * PSI_F + n * PSI_F + k];
    }
    float rc = 1.0f / (float)cn;
    #pragma unroll
    for (int k = 0; k < PSI_F; ++k) {
        float zk = ps[k] * rc;
        #pragma unroll
        for (int j = 0; j < FH; ++j)
            g[j] = fmaf(zk, sW1[(IN_F + k) * FH + j], g[j]);
    }
    #pragma unroll
    for (int j = 0; j < FH; ++j) sG[t * (FH + 1) + j] = fmaxf(g[j], 0.0f);
    __syncthreads();

    // coalesced output: thread t produces out columns t and t+64 for 64 nodes
    for (int nl = 0; nl < 64; ++nl) {
        float o0 = sB2[t], o1 = sB2[t + 64];
        #pragma unroll
        for (int j = 0; j < FH; ++j) {
            float gj = sG[nl * (FH + 1) + j];
            o0 = fmaf(gj, sW2[j * OUT_F + t], o0);
            o1 = fmaf(gj, sW2[j * OUT_F + t + 64], o1);
        }
        out[(n0 + nl) * OUT_F + t]      = fmaxf(o0, 0.0f);
        out[(n0 + nl) * OUT_F + t + 64] = fmaxf(o1, 0.0f);
    }
}

extern "C" void kernel_launch(void* const* d_in, const int* in_sizes, int n_in,
                              void* d_out, int out_size, void* d_ws, size_t ws_size,
                              hipStream_t stream) {
    const float* nodes  = (const float*)d_in[0];
    const int*   adj    = (const int*)d_in[1];
    const float* psi_w1 = (const float*)d_in[2];
    const float* psi_b1 = (const float*)d_in[3];
    const float* psi_w2 = (const float*)d_in[4];
    const float* psi_b2 = (const float*)d_in[5];
    const float* fi_w1  = (const float*)d_in[6];
    const float* fi_b1  = (const float*)d_in[7];
    const float* fi_w2  = (const float*)d_in[8];
    const float* fi_b2  = (const float*)d_in[9];
    float* out = (float*)d_out;

    char* ws = (char*)d_ws;
    unsigned short* hwB      = (unsigned short*)ws;                    // 512 KB
    float*          psi_part = (float*)(ws + (size_t)512 * 1024);      // 8 MB
    unsigned int*   cnt_part =
        (unsigned int*)(ws + (size_t)(512 + 8192) * 1024);             // 512 KB

    psi_mlp_kernel<<<NNODES / 64, 64, 0, stream>>>(
        nodes, psi_w1, psi_b1, psi_w2, psi_b2, hwB);
    aggregate_kernel<<<dim3(NNODES / 32, SLICES), 64, 0, stream>>>(
        adj, (const unsigned int*)hwB, psi_part, cnt_part);
    fi_mlp_kernel<<<NNODES / 64, 64, 0, stream>>>(
        nodes, psi_part, cnt_part, fi_w1, fi_b1, fi_w2, fi_b2, out);
}
